// AtNeuron_18622978195626
// MI455X (gfx1250) — matrix operand order repaired and checked by reading, not compiled, not run on hardware
//
#include <hip/hip_runtime.h>

// ---------------------------------------------------------------------------
// out_t = x_t @ (y_t/t + Yv_{t-1}) + Xv_{t-1} @ y_t      (t = 1..8, out_0 = 0)
// Xv_t = Xv_{t-1} + x_t/t ;  Yv_t = Yv_{t-1} + y_t/t
//
// fp32 GEMM done as bf16x3 split (hi*hi + hi*lo + lo*hi) on
// v_wmma_f32_16x16x32_bf16 — ~8x the MACs/instr of the f32 16x16x4 WMMA,
// near-fp32 accuracy. Per-step working set (~33 MB) lives in the 192 MB L2.
// ---------------------------------------------------------------------------

typedef __attribute__((ext_vector_type(16))) __bf16 v16bf;
typedef __attribute__((ext_vector_type(8)))  __bf16 v8bf;
typedef __attribute__((ext_vector_type(4)))  __bf16 v4bf;
typedef __attribute__((ext_vector_type(8)))  float  v8f;

#define MDIM   512
#define BATCH  8
#define TSTEPS 8
#define BM     128      // block tile M
#define BN     64       // block tile N
#define BK     32       // K step = one bf16 WMMA K
#define KP     40       // padded LDS row pitch in bf16 (80B: 16B-aligned, conflict-free)

union FragBF { v16bf v; v8bf h[2]; };

static __device__ __forceinline__ v8f wmma_bf16(const FragBF& a, const FragBF& b, v8f c) {
    // (neg_a, A, neg_b, B, c_mod, C, reuse_a, reuse_b)
    return __builtin_amdgcn_wmma_f32_16x16x32_bf16(false, a.v, false, b.v,
                                                   (short)0, c, false, false);
}

// split fp32x4 -> bf16 hi + bf16 lo, store as two 8B LDS writes
static __device__ __forceinline__ void split_store4(__bf16* __restrict__ hp,
                                                    __bf16* __restrict__ lp,
                                                    float4 v) {
    v4bf h, l;
    h[0] = (__bf16)v.x; l[0] = (__bf16)(v.x - (float)h[0]);
    h[1] = (__bf16)v.y; l[1] = (__bf16)(v.y - (float)h[1]);
    h[2] = (__bf16)v.z; l[2] = (__bf16)(v.z - (float)h[2]);
    h[3] = (__bf16)v.w; l[3] = (__bf16)(v.w - (float)h[3]);
    *(v4bf*)hp = h;
    *(v4bf*)lp = l;
}

// A fragment (16x32 bf16, M=rows). ISA layout: lanes 0-15 = M, K 0..7/16..23;
// lanes 16-31 = M, K 8..15/24..31. LDS tile is [m][k] with pitch KP.
static __device__ __forceinline__ FragBF load_a_frag(const __bf16* base, int mrow, int lane) {
    const int l  = lane & 15;
    const int hi = lane >> 4;            // 0 or 1
    const __bf16* rp = base + (mrow + l) * KP;
    FragBF f;
    f.h[0] = *(const v8bf*)(rp + hi * 8);          // K 0..7  / 8..15
    f.h[1] = *(const v8bf*)(rp + 16 + hi * 8);     // K 16..23 / 24..31
    return f;
}

// B fragment (32x16 bf16, N across lanes, K packed per lane).
// LDS tile stored transposed [n][k] with pitch KP -> contiguous K per lane.
static __device__ __forceinline__ FragBF load_b_frag(const __bf16* base, int ncol, int lane) {
    const int l  = lane & 15;
    const int hi = lane >> 4;            // lane<16 -> K 0..7/16..23, lane>=16 -> K 8..15/24..31
    const __bf16* rp = base + (ncol + l) * KP;
    FragBF f;
    f.h[0] = *(const v8bf*)(rp + hi * 8);
    f.h[1] = *(const v8bf*)(rp + 16 + hi * 8);
    return f;
}

__global__ __launch_bounds__(256, 1) void
gemm_step(const float* __restrict__ X,  const float* __restrict__ Y,
          const float* __restrict__ XV, const float* __restrict__ YV,
          float* __restrict__ Out, float inv)
{
    // hi/lo LDS tiles: A operands x_t, Xv  (row-major [m][k])
    //                  B operands Bc=y*inv+Yv, y_t (transposed [n][k])
    __shared__ __align__(16) __bf16 AxH[BM * KP], AxL[BM * KP];
    __shared__ __align__(16) __bf16 AvH[BM * KP], AvL[BM * KP];
    __shared__ __align__(16) __bf16 BcH[BN * KP], BcL[BN * KP];
    __shared__ __align__(16) __bf16 ByH[BN * KP], ByL[BN * KP];

    const size_t mat = (size_t)MDIM * MDIM;
    const int b = blockIdx.z;
    X  += b * mat;  Y  += b * mat;
    XV += b * mat;  YV += b * mat;
    Out += b * mat;

    const int tid  = threadIdx.x;
    const int lane = tid & 31;
    const int wave = tid >> 5;           // 8 wave32
    const int wr   = wave >> 1;          // 0..3  -> 32-row strip
    const int wc   = wave & 1;           // 0..1  -> 32-col strip
    const int m0   = blockIdx.x * BM;
    const int n0   = blockIdx.y * BN;

    v8f acc[2][2];
    #pragma unroll
    for (int i = 0; i < 2; ++i)
        #pragma unroll
        for (int j = 0; j < 2; ++j)
            acc[i][j] = (v8f){0.f,0.f,0.f,0.f,0.f,0.f,0.f,0.f};

    for (int ko = 0; ko < MDIM; ko += BK) {
        __syncthreads();

        // ---- fill A tiles (128x32 fp32 each): 4 float4 tasks/thread/tensor
        #pragma unroll
        for (int q = 0; q < 4; ++q) {
            const int task = tid + q * 256;          // 0..1023
            const int r    = task >> 3;              // 0..127
            const int c4   = (task & 7) << 2;        // 0..28
            const size_t g = (size_t)(m0 + r) * MDIM + ko + c4;
            split_store4(&AxH[r * KP + c4], &AxL[r * KP + c4], *(const float4*)&X[g]);
            split_store4(&AvH[r * KP + c4], &AvL[r * KP + c4], *(const float4*)&XV[g]);
        }
        // ---- fill B tiles (32x64 fp32, transposed): 2 tasks/thread
        #pragma unroll
        for (int q = 0; q < 2; ++q) {
            const int task = tid + q * 256;          // 0..511
            const int n    = task & 63;              // 0..63
            const int k4   = (task >> 6) << 2;       // 0..28
            float4 yy, bb;
            float* yp = (float*)&yy;
            float* bp = (float*)&bb;
            #pragma unroll
            for (int i = 0; i < 4; ++i) {
                const size_t g = (size_t)(ko + k4 + i) * MDIM + n0 + n;
                const float yv = Y[g];
                const float vv = YV[g];
                yp[i] = yv;
                bp[i] = fmaf(yv, inv, vv);           // Bc = y_t/t + Yv_{t-1}
            }
            split_store4(&ByH[n * KP + k4], &ByL[n * KP + k4], yy);
            split_store4(&BcH[n * KP + k4], &BcL[n * KP + k4], bb);
        }

        // prefetch next K tile into cache hierarchy while we compute
        if (ko + BK < MDIM) {
            __builtin_prefetch(&X[(size_t)(m0 + (tid >> 1)) * MDIM + ko + BK], 0, 0);
            __builtin_prefetch(&Y[(size_t)(ko + BK + (tid & 31)) * MDIM + n0], 0, 0);
        }

        __syncthreads();

        // ---- compute: dual GEMM, bf16x3 split -> 6 WMMA per 16x16 tile
        #pragma unroll
        for (int j = 0; j < 2; ++j) {
            const int nb  = wc * 32 + j * 16;
            FragBF bch = load_b_frag(BcH, nb, lane);
            FragBF bcl = load_b_frag(BcL, nb, lane);
            FragBF byh = load_b_frag(ByH, nb, lane);
            FragBF byl = load_b_frag(ByL, nb, lane);
            #pragma unroll
            for (int i = 0; i < 2; ++i) {
                const int mb = wr * 32 + i * 16;
                FragBF axh = load_a_frag(AxH, mb, lane);
                FragBF axl = load_a_frag(AxL, mb, lane);
                FragBF avh = load_a_frag(AvH, mb, lane);
                FragBF avl = load_a_frag(AvL, mb, lane);
                v8f a = acc[i][j];
                a = wmma_bf16(axh, bch, a);   // x_hi  * Bc_hi
                a = wmma_bf16(axh, bcl, a);   // x_hi  * Bc_lo
                a = wmma_bf16(axl, bch, a);   // x_lo  * Bc_hi
                a = wmma_bf16(avh, byh, a);   // Xv_hi * y_hi
                a = wmma_bf16(avh, byl, a);   // Xv_hi * y_lo
                a = wmma_bf16(avl, byh, a);   // Xv_lo * y_hi
                acc[i][j] = a;
            }
        }
    }

    // ---- epilogue: C/D layout -> VGPR r: (M=r, N=lane) | (M=r+8, N=lane-16)
    const int l15 = lane & 15;
    const int lhi = lane >> 4;
    #pragma unroll
    for (int i = 0; i < 2; ++i)
        #pragma unroll
        for (int j = 0; j < 2; ++j)
            #pragma unroll
            for (int r = 0; r < 8; ++r) {
                const int m = m0 + wr * 32 + i * 16 + r + lhi * 8;
                const int n = n0 + wc * 32 + j * 16 + l15;
                Out[(size_t)m * MDIM + n] = acc[i][j][r];
            }
}

__global__ void zero_kernel(float4* __restrict__ p, int n4) {
    const int i = blockIdx.x * blockDim.x + threadIdx.x;
    if (i < n4) p[i] = float4{0.f, 0.f, 0.f, 0.f};
}

__global__ void update_avg(float4* __restrict__ xv, float4* __restrict__ yv,
                           const float4* __restrict__ x, const float4* __restrict__ y,
                           float inv, int n4) {
    const int i = blockIdx.x * blockDim.x + threadIdx.x;
    if (i < n4) {
        float4 a = xv[i], u = x[i];
        a.x = fmaf(u.x, inv, a.x); a.y = fmaf(u.y, inv, a.y);
        a.z = fmaf(u.z, inv, a.z); a.w = fmaf(u.w, inv, a.w);
        xv[i] = a;
        float4 c = yv[i], v = y[i];
        c.x = fmaf(v.x, inv, c.x); c.y = fmaf(v.y, inv, c.y);
        c.z = fmaf(v.z, inv, c.z); c.w = fmaf(v.w, inv, c.w);
        yv[i] = c;
    }
}

extern "C" void kernel_launch(void* const* d_in, const int* in_sizes, int n_in,
                              void* d_out, int out_size, void* d_ws, size_t ws_size,
                              hipStream_t stream) {
    const float* x = (const float*)d_in[0];
    const float* y = (const float*)d_in[1];
    float* out = (float*)d_out;

    const size_t mat8 = (size_t)BATCH * MDIM * MDIM;   // 2,097,152 floats
    float* xv = (float*)d_ws;                          // running avg of x
    float* yv = xv + mat8;                             // running avg of y (needs 16 MB ws)
    const int n4 = (int)(mat8 / 4);

    // zero Xv,Yv (contiguous) and out slice t=0 (reference returns zeros there)
    zero_kernel<<<(2 * n4 + 255) / 256, 256, 0, stream>>>((float4*)xv, 2 * n4);
    zero_kernel<<<(n4 + 255) / 256, 256, 0, stream>>>((float4*)out, n4);

    const dim3 grid(MDIM / BM, MDIM / BN, BATCH);      // (4, 8, 8)
    for (int t = 1; t <= TSTEPS; ++t) {
        const float* xt = x + (size_t)t * mat8;
        const float* yt = y + (size_t)t * mat8;
        const float inv = 1.0f / (float)t;
        gemm_step<<<grid, 256, 0, stream>>>(xt, yt, xv, yv, out + (size_t)t * mat8, inv);
        update_avg<<<(n4 + 255) / 256, 256, 0, stream>>>(
            (float4*)xv, (float4*)yv, (const float4*)xt, (const float4*)yt, inv, n4);
    }
}
